// MemoryModule_50912542327268
// MI455X (gfx1250) — matrix operand order repaired and checked by reading, not compiled, not run on hardware
//
#include <hip/hip_runtime.h>

typedef float    v8f  __attribute__((ext_vector_type(8)));
typedef _Float16 v16h __attribute__((ext_vector_type(16)));
typedef _Float16 v8h  __attribute__((ext_vector_type(8)));

union V16 { v16h v; v8h h[2]; };

#define T_ROWS 32768
#define CDIM   512
#define MDIM   512

__device__ __forceinline__ float wave_sum(float v) {
  #pragma unroll
  for (int off = 16; off > 0; off >>= 1) v += __shfl_xor(v, off, 32);
  return v;
}

__device__ __forceinline__ void wave_argmax(float& m, int& a) {
  #pragma unroll
  for (int off = 16; off > 0; off >>= 1) {
    float om = __shfl_xor(m, off, 32);
    int   oa = __shfl_xor(a, off, 32);
    if (om > m || (om == m && oa < a)) { m = om; a = oa; }
  }
}

// K0: keys -> f16 (row-major [M,C]); values -> f16 transposed [C,M]
__global__ __launch_bounds__(256) void convert_kernel(
    const float* __restrict__ keys, const float* __restrict__ values,
    _Float16* __restrict__ keysf, _Float16* __restrict__ valT) {
  const int i = blockIdx.x * 256 + threadIdx.x;   // 0 .. 262143
  keysf[i] = (_Float16)keys[i];
  const int m = i >> 9, c = i & 511;
  valT[(size_t)c * MDIM + m] = (_Float16)values[i];
}

// K1: L2-normalize key rows -> f16. One wave per row, 8 rows per block.
__global__ __launch_bounds__(256) void normalize_kernel(
    const float* __restrict__ x, _Float16* __restrict__ y) {
  const int wave = threadIdx.x >> 5, lane = threadIdx.x & 31;
  const size_t row = (size_t)blockIdx.x * 8 + wave;
  const float* xp = x + row * CDIM;
  float v[16], ss = 0.f;
  #pragma unroll
  for (int j = 0; j < 16; ++j) { v[j] = xp[lane + 32 * j]; ss += v[j] * v[j]; }
  ss = wave_sum(ss);
  const float s = 1.0f / fmaxf(sqrtf(ss), 1e-12f);
  _Float16* yp = y + row * CDIM;
  #pragma unroll
  for (int j = 0; j < 16; ++j) yp[lane + 32 * j] = (_Float16)(v[j] * s);
}

// K2: fused score GEMM + argmax + softmax/entropy/contrast/gathering.
// Block = 16 token rows x full M=512. 8 waves, each wave a 16x64 WMMA tile.
__global__ __launch_bounds__(256) void score_kernel(
    const _Float16* __restrict__ kn,    // [T,C] f16, normalized keys
    const _Float16* __restrict__ keysf, // [M,C] f16
    const float* __restrict__ query,    // [T,C] f32
    const float* __restrict__ values,   // [M,C] f32
    _Float16* __restrict__ wts,         // [T,M] f16 softmax weights (out)
    float* __restrict__ partials) {     // [grid,3] (entropy,gathering,contrast)
  __shared__ float scr[16][MDIM];       // 32 KB score/exp tile
  __shared__ float pE[8], pG[8], pC[8];
  const int wave = threadIdx.x >> 5, lane = threadIdx.x & 31;
  const int rowbase = blockIdx.x * 16;
  const int arow = lane & 15;
  const int ksel = (lane & 16) ? 8 : 0;   // A: two 8-half runs per lane
  const int bsel = (lane & 16) ? 16 : 0;  // B: one 16-half run per lane
  const int rsel = (lane & 16) ? 8 : 0;   // C/D: row offset
  const int n0 = wave * 64;

  v8f acc[4];
  const v8f vzero = {0.f,0.f,0.f,0.f,0.f,0.f,0.f,0.f};
  #pragma unroll
  for (int t = 0; t < 4; ++t) acc[t] = vzero;

  const _Float16* ap = kn + (size_t)(rowbase + arow) * CDIM;
  for (int kb = 0; kb < CDIM; kb += 32) {
    V16 a;
    a.h[0] = *(const v8h*)(ap + kb + 2 * ksel);
    a.h[1] = *(const v8h*)(ap + kb + 2 * ksel + 8);
    #pragma unroll
    for (int t = 0; t < 4; ++t) {
      const int n = n0 + t * 16 + arow;   // slot index m
      const v16h b = *(const v16h*)(keysf + (size_t)n * CDIM + kb + bsel);
      acc[t] = __builtin_amdgcn_wmma_f32_16x16x32_f16(
          false, a.v, false, b, (short)0, acc[t], false, false);
    }
  }
  #pragma unroll
  for (int t = 0; t < 4; ++t)
    #pragma unroll
    for (int v = 0; v < 8; ++v)
      scr[v + rsel][n0 + t * 16 + arow] = acc[t][v];
  __syncthreads();

  float wE = 0.f, wG = 0.f, wC = 0.f;
  for (int rr = 0; rr < 2; ++rr) {
    const int r = wave + rr * 8;
    const size_t trow = (size_t)(rowbase + r);
    // pass 1: max + first-argmax
    float m = -3.4e38f; int arg = 0;
    #pragma unroll
    for (int j = 0; j < 16; ++j) {
      const int c = lane + 32 * j;
      const float v = scr[r][c];
      if (v > m) { m = v; arg = c; }
    }
    wave_argmax(m, arg);
    // pass 2: exp sums; s2 = sum exp(2(s-m)) for T=0.5 log-softmax
    float s1 = 0.f, s2 = 0.f;
    #pragma unroll
    for (int j = 0; j < 16; ++j) {
      const int c = lane + 32 * j;
      const float e = expf(scr[r][c] - m);
      s1 += e; s2 += e * e;
      scr[r][c] = e;
    }
    s1 = wave_sum(s1); s2 = wave_sum(s2);
    const float invZ = 1.0f / s1;
    // pass 3: weights out + entropy
    float H = 0.f;
    _Float16* wp = wts + trow * MDIM;
    #pragma unroll
    for (int j = 0; j < 16; ++j) {
      const int c = lane + 32 * j;
      const float w = scr[r][c] * invZ;
      H -= w * logf(w + 1e-12f);
      wp[c] = (_Float16)w;
    }
    H = wave_sum(H);
    // gathering loss: normalize q row on the fly, compare to values[arg]
    const float* qp = query + trow * CDIM;
    float qv[16], ss = 0.f;
    #pragma unroll
    for (int j = 0; j < 16; ++j) { qv[j] = qp[lane + 32 * j]; ss += qv[j] * qv[j]; }
    ss = wave_sum(ss);
    const float scale = 1.0f / fmaxf(sqrtf(ss), 1e-12f);
    const float* vp = values + (size_t)arg * CDIM;
    float g = 0.f;
    #pragma unroll
    for (int j = 0; j < 16; ++j) {
      const float d = qv[j] * scale - vp[lane + 32 * j];
      g += d * d;
    }
    g = wave_sum(g);
    wE += H * (1.0f / 32768.0f);
    wG += g * (1.0f / (32768.0f * 512.0f));
    // gathered = -log(s2); contrast = sum(gathered)/(-12800) => +log(s2)/12800
    wC += logf(s2) * (1.0f / 12800.0f);
  }
  if (lane == 0) { pE[wave] = wE; pG[wave] = wG; pC[wave] = wC; }
  __syncthreads();
  if (threadIdx.x == 0) {
    float e = 0.f, g = 0.f, c = 0.f;
    for (int i = 0; i < 8; ++i) { e += pE[i]; g += pG[i]; c += pC[i]; }
    partials[blockIdx.x * 3 + 0] = e;
    partials[blockIdx.x * 3 + 1] = g;
    partials[blockIdx.x * 3 + 2] = c;
  }
}

// K3: read_query = weights[T,M] @ values[M,C] via f16 WMMA (values pre-transposed [C,M])
__global__ __launch_bounds__(256) void read_gemm_kernel(
    const _Float16* __restrict__ wts, const _Float16* __restrict__ valT,
    float* __restrict__ out) {
  const int wave = threadIdx.x >> 5, lane = threadIdx.x & 31;
  const int rowbase = blockIdx.x * 16;
  const int arow = lane & 15;
  const int ksel = (lane & 16) ? 8 : 0;
  const int bsel = (lane & 16) ? 16 : 0;
  const int rsel = (lane & 16) ? 8 : 0;
  const int n0 = wave * 64;

  v8f acc[4];
  const v8f vzero = {0.f,0.f,0.f,0.f,0.f,0.f,0.f,0.f};
  #pragma unroll
  for (int t = 0; t < 4; ++t) acc[t] = vzero;

  const _Float16* ap = wts + (size_t)(rowbase + arow) * MDIM;
  for (int kb = 0; kb < MDIM; kb += 32) {
    V16 a;
    a.h[0] = *(const v8h*)(ap + kb + 2 * ksel);
    a.h[1] = *(const v8h*)(ap + kb + 2 * ksel + 8);
    #pragma unroll
    for (int t = 0; t < 4; ++t) {
      const int n = n0 + t * 16 + arow;   // output channel c
      const v16h b = *(const v16h*)(valT + (size_t)n * MDIM + kb + bsel);
      acc[t] = __builtin_amdgcn_wmma_f32_16x16x32_f16(
          false, a.v, false, b, (short)0, acc[t], false, false);
    }
  }
  #pragma unroll
  for (int t = 0; t < 4; ++t)
    #pragma unroll
    for (int v = 0; v < 8; ++v)
      out[(size_t)(rowbase + v + rsel) * CDIM + n0 + t * 16 + arow] = acc[t][v];
}

// K4: deterministic fixed-order reduce of 2048 partial triples -> 3 scalars
__global__ void final_reduce_kernel(const float* __restrict__ partials, int nblk,
                                    float* __restrict__ out_scalars) {
  const int lane = threadIdx.x;  // 32 threads
  float e = 0.f, g = 0.f, c = 0.f;
  for (int i = lane; i < nblk; i += 32) {
    e += partials[i * 3 + 0];
    g += partials[i * 3 + 1];
    c += partials[i * 3 + 2];
  }
  e = wave_sum(e); g = wave_sum(g); c = wave_sum(c);
  if (lane == 0) { out_scalars[0] = e; out_scalars[1] = g; out_scalars[2] = c; }
}

extern "C" void kernel_launch(void* const* d_in, const int* in_sizes, int n_in,
                              void* d_out, int out_size, void* d_ws, size_t ws_size,
                              hipStream_t stream) {
  const float* key    = (const float*)d_in[0];
  const float* query  = (const float*)d_in[1];
  const float* keys   = (const float*)d_in[2];
  const float* values = (const float*)d_in[3];
  float* out = (float*)d_out;

  // workspace layout
  char* ws = (char*)d_ws;
  _Float16* knorm = (_Float16*)(ws);                       // 32 MB
  _Float16* wts   = (_Float16*)(ws + (size_t)33554432);    // 32 MB
  _Float16* keysf = (_Float16*)(ws + (size_t)67108864);    // 0.5 MB
  _Float16* valT  = (_Float16*)(ws + (size_t)67633152);    // 0.5 MB
  float*    parts = (float*)   (ws + (size_t)68157440);    // 24 KB

  // output offsets (floats): read_query | entropy | gathering | contrast | keys | values
  const size_t OFF_SCALARS = (size_t)T_ROWS * CDIM;        // 16777216
  const size_t OFF_KEYS    = OFF_SCALARS + 3;              // 16777219
  const size_t OFF_VALS    = OFF_KEYS + (size_t)MDIM * CDIM;

  convert_kernel   <<<1024, 256, 0, stream>>>(keys, values, keysf, valT);
  normalize_kernel <<<T_ROWS / 8, 256, 0, stream>>>(key, knorm);
  score_kernel     <<<T_ROWS / 16, 256, 0, stream>>>(knorm, keysf, query, values, wts, parts);
  read_gemm_kernel <<<T_ROWS / 16, 256, 0, stream>>>(wts, valT, out);
  final_reduce_kernel<<<1, 32, 0, stream>>>(parts, T_ROWS / 16, out + OFF_SCALARS);

  hipMemcpyAsync(out + OFF_KEYS, keys,   (size_t)MDIM * CDIM * sizeof(float),
                 hipMemcpyDeviceToDevice, stream);
  hipMemcpyAsync(out + OFF_VALS, values, (size_t)MDIM * CDIM * sizeof(float),
                 hipMemcpyDeviceToDevice, stream);
}
